// CRFLoss_62234076119722
// MI455X (gfx1250) — matrix operand order repaired and checked by reading, not compiled, not run on hardware
//
#include <hip/hip_runtime.h>

// ---------------------------------------------------------------------------
// Dense-CRF pairwise loss on gfx1250 (wave32, WMMA).
//   loss = mean_b [ sum_c S_c^T K (1 - S_c) / N ]
//   K_ij = exp(-|dxy|^2/(2*15^2) - |drgb|^2/(2*0.125^2))
//
// The WMMA output IS the base-2 exponent: feature vectors carry both the
// (hi/lo split) scaled pixel features and the (hi/lo split) squared norms,
// so D[r] = 2*log2e*f_i.f_j - log2e*|f_i|^2 - log2e*|f_j|^2 and the
// per-element epilogue is a single raw v_exp_f32 plus the class weights.
// ---------------------------------------------------------------------------

typedef __attribute__((ext_vector_type(16))) _Float16 v16h;
typedef __attribute__((ext_vector_type(8)))  float    v8f;
typedef __attribute__((ext_vector_type(4)))  float    f32x4;

#define HH    96
#define WW    96
#define NPIX  (HH * WW)      // 9216
#define TILE  16
#define NT    (NPIX / TILE)  // 576 tiles per dimension
#define WAVES 8              // 256-thread block = 8 wave32
#define LOG2E 1.4426950408889634f

#if __has_builtin(__builtin_amdgcn_exp2f)
#define RAW_EXP2(x) __builtin_amdgcn_exp2f(x)   // bare v_exp_f32
#else
#define RAW_EXP2(x) exp2f(x)
#endif

// sqrt(1/(2*sigma^2)) feature scales
__device__ __constant__ float kSXY  = 0.047140452079103168f;  // 1/(sqrt(2)*15)
__device__ __constant__ float kSRGB = 5.656854249492381f;     // 1/(sqrt(2)*0.125)

// ---------------------------------------------------------------------------
// Kernel 1a: packed WMMA operand rows (A-side and B-side) per pixel.
//  u = (2*log2e)*f  (hi/lo split),  v = f (hi/lo split),  m = log2e*|f|^2
//  A[i][32]: 0-4 u_hi, 5-9 u_lo, 10-14 u_hi, 15 -m_hi, 16 -m_lo, 17 -1, 18 -1
//  B[j][32]: 0-4 v_hi, 5-9 v_hi, 10-14 v_lo, 15  1,    16  1,    17 m_hi, 18 m_lo
//  => (A row_i).(B col_j) = 2*log2e*f_i.f_j - m_i - m_j  (the base-2 exponent)
// ---------------------------------------------------------------------------
__global__ void crf_precompute(const float* __restrict__ image,
                               _Float16* __restrict__ Afeat,
                               _Float16* __restrict__ Bfeat,
                               int B) {
    int idx = blockIdx.x * blockDim.x + threadIdx.x;
    if (idx >= B * NPIX) return;
    int bb = idx / NPIX;
    int i  = idx - bb * NPIX;
    int y  = i / WW;
    int x  = i - y * WW;

    float f[5];
    f[0] = (float)y * kSXY;
    f[1] = (float)x * kSXY;
    f[2] = image[((size_t)bb * 3 + 0) * NPIX + i] * kSRGB;
    f[3] = image[((size_t)bb * 3 + 1) * NPIX + i] * kSRGB;
    f[4] = image[((size_t)bb * 3 + 2) * NPIX + i] * kSRGB;

    float n = 0.0f;
#pragma unroll
    for (int k = 0; k < 5; ++k) n = __builtin_fmaf(f[k], f[k], n);
    const float m = n * LOG2E;

    _Float16* A  = Afeat + (size_t)idx * 32;
    _Float16* Bp = Bfeat + (size_t)idx * 32;
#pragma unroll
    for (int k = 0; k < 32; ++k) { A[k] = (_Float16)0.0f; Bp[k] = (_Float16)0.0f; }

#pragma unroll
    for (int k = 0; k < 5; ++k) {
        const float u = f[k] * (2.0f * LOG2E);
        const _Float16 uhi = (_Float16)u;
        const _Float16 ulo = (_Float16)(u - (float)uhi);
        const _Float16 vhi = (_Float16)f[k];
        const _Float16 vlo = (_Float16)(f[k] - (float)vhi);
        A[k]      = uhi;  Bp[k]      = vhi;
        A[5 + k]  = ulo;  Bp[5 + k]  = vhi;
        A[10 + k] = uhi;  Bp[10 + k] = vlo;
    }
    const _Float16 mhi = (_Float16)m;
    const _Float16 mlo = (_Float16)(m - (float)mhi);
    A[15] = -mhi;           Bp[15] = (_Float16)1.0f;
    A[16] = -mlo;           Bp[16] = (_Float16)1.0f;
    A[17] = (_Float16)-1.0f; Bp[17] = mhi;
    A[18] = (_Float16)-1.0f; Bp[18] = mlo;
}

// Kernel 1b: q = 1 - p, precomputed so the j-loop has no subtracts.
__global__ void crf_prep_q(const float* __restrict__ probs,
                           float* __restrict__ Q, int total) {
    int idx = blockIdx.x * blockDim.x + threadIdx.x;
    if (idx < total) Q[idx] = 1.0f - probs[idx];
}

// ---------------------------------------------------------------------------
// Kernel 2: main N^2 sweep. One block per i-tile (grid.x = 576, grid.y = B).
// Each wave32 holds the A fragment + row probs in VGPRs and iterates j-tiles
// with stride 8.  Per tile: 1 WMMA + 8 x (v_exp + CT+1 FMAs) per lane.
// CT = #classes handled this pass (template-specialized), c0 = class offset.
// ---------------------------------------------------------------------------
template <int CT>
__global__ void __launch_bounds__(256)
crf_main(const float* __restrict__ probs,
         const _Float16* __restrict__ Afeat,
         const _Float16* __restrict__ Bfeat,
         const float* __restrict__ Q,
         float* __restrict__ partials,
         int C, int c0, int pass) {
    const int it   = blockIdx.x;            // i-tile index
    const int bb   = blockIdx.y;            // batch index
    const int lane = threadIdx.x & 31;
    const int wave = threadIdx.x >> 5;
    const int hi   = lane >> 4;             // lane group 0/1
    const int l16  = lane & 15;
    const int i0   = it * TILE;

    // ---- A fragment (16x32 f16): lanes 0-15 rows M=0..15 K[0..7|16..23],
    //      lanes 16-31 same rows K[8..15|24..31]   (ISA 7.12.2 wave32 layout)
    union { v16h h; f32x4 f4[2]; } afrag;
    {
        const int irow = i0 + l16;
        const f32x4* ap = (const f32x4*)(Afeat + ((size_t)bb * NPIX + irow) * 32);
        afrag.f4[0] = ap[hi];
        afrag.f4[1] = ap[2 + hi];
    }

    // ---- row-side probs: C/D row M = r + hi*8, column = l16
    float prow[CT][8];
#pragma unroll
    for (int r = 0; r < 8; ++r) {
        const int ir = i0 + r + hi * 8;
#pragma unroll
        for (int c = 0; c < CT; ++c)
            prow[c][r] = probs[((size_t)bb * C + (c0 + c)) * NPIX + ir];
    }

    float acc = 0.0f;
    for (int jt = wave; jt < NT; jt += WAVES) {
        const int j = jt * TILE + l16;      // this lane's column

        // ---- B fragment (32x16 f16): column = l16,
        //      lanes 0-15 K[0..7|16..23], lanes 16-31 K[8..15|24..31] (matches A)
        union { v16h h; f32x4 f4[2]; } bfrag;
        const f32x4* bp = (const f32x4*)(Bfeat + ((size_t)bb * NPIX + j) * 32);
        bfrag.f4[0] = bp[hi];
        bfrag.f4[1] = bp[2 + hi];

        float qcol[CT];
#pragma unroll
        for (int c = 0; c < CT; ++c)
            qcol[c] = Q[((size_t)bb * C + (c0 + c)) * NPIX + j];

        v8f cz = {0.f, 0.f, 0.f, 0.f, 0.f, 0.f, 0.f, 0.f};
        // D[r] = base-2 exponent of the bilateral kernel (norms folded in)
        v8f e2 = __builtin_amdgcn_wmma_f32_16x16x32_f16(
            /*neg_a=*/false, afrag.h, /*neg_b=*/false, bfrag.h,
            /*c_mod=*/(short)0, cz, /*reuse_a=*/false, /*reuse_b=*/false);

#pragma unroll
        for (int r = 0; r < 8; ++r) {
            const float k = RAW_EXP2(e2[r]);   // bare v_exp_f32
            float w = 0.0f;
#pragma unroll
            for (int c = 0; c < CT; ++c)
                w = __builtin_fmaf(prow[c][r], qcol[c], w);
            acc = __builtin_fmaf(k, w, acc);
        }
    }

    // ---- deterministic block tree reduction
    __shared__ float red[256];
    red[threadIdx.x] = acc;
    __syncthreads();
#pragma unroll
    for (int s = 128; s > 0; s >>= 1) {
        if ((int)threadIdx.x < s) red[threadIdx.x] += red[threadIdx.x + s];
        __syncthreads();
    }
    if (threadIdx.x == 0)
        partials[((size_t)pass * gridDim.y + bb) * NT + it] = red[0];
}

// ---------------------------------------------------------------------------
// Kernel 3: fixed-order final reduction (deterministic, no atomics)
// ---------------------------------------------------------------------------
__global__ void crf_finalize(const float* __restrict__ partials,
                             float* __restrict__ out, int nPartials, int B) {
    __shared__ float red[256];
    float s = 0.0f;
    for (int i = threadIdx.x; i < nPartials; i += 256) s += partials[i];
    red[threadIdx.x] = s;
    __syncthreads();
#pragma unroll
    for (int t = 128; t > 0; t >>= 1) {
        if ((int)threadIdx.x < t) red[threadIdx.x] += red[threadIdx.x + t];
        __syncthreads();
    }
    if (threadIdx.x == 0)
        out[0] = red[0] * (1.0f / ((float)NPIX * (float)B));
}

// ---------------------------------------------------------------------------
extern "C" void kernel_launch(void* const* d_in, const int* in_sizes, int n_in,
                              void* d_out, int out_size, void* d_ws, size_t ws_size,
                              hipStream_t stream) {
    const float* probs = (const float*)d_in[0];   // [B, C, H, W] f32
    const float* image = (const float*)d_in[1];   // [B, 3, H, W] f32

    int B = in_sizes[1] / (3 * NPIX);
    if (B < 1) B = 1;
    int C = in_sizes[0] / (B * NPIX);
    if (C < 1) C = 1;
    const int passes = (C + 3) / 4;               // <=4 classes per pass

    // workspace carve-up (f16 = 2 bytes)
    char* ws = (char*)d_ws;
    size_t szFeat = (size_t)B * NPIX * 32 * 2;
    size_t offA = 0;
    size_t offB = (offA + szFeat + 255) & ~(size_t)255;
    size_t offQ = (offB + szFeat + 255) & ~(size_t)255;
    size_t szQ  = (size_t)B * C * NPIX * sizeof(float);
    size_t offP = (offQ + szQ + 255) & ~(size_t)255;

    _Float16* Af = (_Float16*)(ws + offA);
    _Float16* Bf = (_Float16*)(ws + offB);
    float*    Qp = (float*)(ws + offQ);
    float*    par = (float*)(ws + offP);

    int tot = B * NPIX;
    crf_precompute<<<(tot + 255) / 256, 256, 0, stream>>>(image, Af, Bf, B);
    int qtot = B * C * NPIX;
    crf_prep_q<<<(qtot + 255) / 256, 256, 0, stream>>>(probs, Qp, qtot);

    dim3 grid(NT, B);
    for (int p = 0; p < passes; ++p) {
        const int c0 = p * 4;
        const int ct = (C - c0 >= 4) ? 4 : (C - c0);
        switch (ct) {
        case 1: crf_main<1><<<grid, 256, 0, stream>>>(probs, Af, Bf, Qp, par, C, c0, p); break;
        case 2: crf_main<2><<<grid, 256, 0, stream>>>(probs, Af, Bf, Qp, par, C, c0, p); break;
        case 3: crf_main<3><<<grid, 256, 0, stream>>>(probs, Af, Bf, Qp, par, C, c0, p); break;
        default: crf_main<4><<<grid, 256, 0, stream>>>(probs, Af, Bf, Qp, par, C, c0, p); break;
        }
    }

    crf_finalize<<<1, 256, 0, stream>>>(par, (float*)d_out, passes * B * NT, B);
}
